// AFNO2DLayer_47717086658614
// MI455X (gfx1250) — matrix operand order repaired and checked by reading, not compiled, not run on hardware
//
#include <hip/hip_runtime.h>
#include <hip/hip_bf16.h>

typedef __attribute__((ext_vector_type(16))) _Float16 v16h;
typedef __attribute__((ext_vector_type(8)))  _Float16 v8h;
typedef __attribute__((ext_vector_type(8)))  float    v8f;

union FragF { v8f v; float f[8]; };
union V8HU  { v8h v; _Float16 h[8]; };

// ---- WMMA wrapper: D = A(16x32 f16) * B(32x16 f16) + C(16x16 f32) ----
__device__ inline v8f wmma(v16h a, v16h b, v8f c) {
    return __builtin_amdgcn_wmma_f32_16x16x32_f16(false, a, false, b, (short)0, c, false, false);
}

// ---- A-fragment from row-major MxK (stride in halfs, 16B-aligned rows) ----
__device__ inline v16h frag_a_load(const _Float16* M, int stride, int row0, int k0, int lane) {
    int r = lane & 15, hi = lane >> 4;
    const _Float16* p = M + (size_t)(row0 + r) * stride + k0 + 8 * hi;
    v8h lo = *(const v8h*)p;
    v8h hh = *(const v8h*)(p + 16);
    return __builtin_shufflevector(lo, hh, 0,1,2,3,4,5,6,7,8,9,10,11,12,13,14,15);
}

// ---- B-fragment from TRANSPOSED [N][K] storage (strideT in halfs, 16B-aligned) ----
__device__ inline v16h frag_b_loadT(const _Float16* MT, int strideT, int k0, int col0, int lane) {
    int n = lane & 15, hi = lane >> 4;
    const _Float16* p = MT + (size_t)(col0 + n) * strideT + k0 + hi * 8;
    v8h lo = *(const v8h*)p;
    v8h hh = *(const v8h*)(p + 16);
    return __builtin_shufflevector(lo, hh, 0,1,2,3,4,5,6,7,8,9,10,11,12,13,14,15);
}

// f16 fragment negation (f16 WMMA has no A/B negate modifier; flip sign bits)
__device__ inline v16h frag_neg(v16h a) {
    union { v16h v; unsigned u[8]; } x; x.v = a;
    #pragma unroll
    for (int i = 0; i < 8; ++i) x.u[i] ^= 0x80008000u;
    return x.v;
}

__device__ inline void nt_h(_Float16* p, _Float16 v) { __builtin_nontemporal_store(v, p); }
__device__ inline void nt_f(float* p, float v)       { __builtin_nontemporal_store(v, p); }

#define LAM 0.01f

// =====================  K0: DFT tables  =====================
__global__ void k_init_tables(_Float16* tcos, _Float16* tsinf, _Float16* tsini,
                              _Float16* icos, _Float16* isin) {
    int idx = blockIdx.x * 256 + threadIdx.x;
    const float inv = 0.08838834764831845f;   // 1/sqrt(128)
    const float w0  = 0.049087385212340517f;  // 2*pi/128
    if (idx < 16384) {
        int k = idx >> 7, nn = idx & 127;
        int m = (k * nn) & 127;               // exact mod-128 angle reduction
        float s = __sinf(w0 * m), c = __cosf(w0 * m);
        tcos[idx]  = (_Float16)(c * inv);
        tsinf[idx] = (_Float16)(-s * inv);
        tsini[idx] = (_Float16)(s * inv);
    } else if (idx < 16384 + 12288) {
        int j = idx - 16384;
        int w = j / 96, k = j % 96;
        float vr = 0.f, vi = 0.f;
        if (k < 65) {
            float a = (k == 0 || k == 64) ? 1.f : 2.f;
            int m = (k * w) & 127;
            vr =  a * __cosf(w0 * m) * inv;
            vi = -a * __sinf(w0 * m) * inv;
        }
        icos[j] = (_Float16)vr;
        isin[j] = (_Float16)vi;
    }
}

// =====================  K1: row rFFT (along W) =====================
__global__ __launch_bounds__(256) void k_row_fft(
    const float* __restrict__ x, const _Float16* __restrict__ tcos,
    const _Float16* __restrict__ tsin,
    _Float16* __restrict__ Yr, _Float16* __restrict__ Yi) {
    __align__(16) __shared__ _Float16 sxT[256 * 136];   // ~68 KB, [c][w]
    int wg = blockIdx.x;                      // 1536 = 512 * 3
    int cg = wg % 3, bh = wg / 3;
    int c0 = cg * 256;
    int tid = threadIdx.x;
    const float* xs = x + (size_t)bh * (128 * 768) + c0;
    for (int j = 0; j < 16; ++j) {            // 4096 b128 stores
        int idx = tid + j * 256;
        int cc = idx & 255, w0 = (idx >> 8) * 8;
        V8HU pk;
        #pragma unroll
        for (int e = 0; e < 8; ++e) pk.h[e] = (_Float16)xs[(size_t)(w0 + e) * 768 + cc];
        *(v8h*)&sxT[cc * 136 + w0] = pk.v;
    }
    __syncthreads();
    int lane = tid & 31, wave = tid >> 5;
    int n = lane & 15, hi = lane >> 4;
    // full tiles: kw tiles 0..3 (kw 0..63), no guard
    for (int t = wave; t < 64; t += 8) {
        int m0 = (t >> 4) << 4, n0 = (t & 15) * 16;
        v8f cr = {}; v8f ci = {};
        #pragma unroll
        for (int kk = 0; kk < 128; kk += 32) {
            v16h ac = frag_a_load(tcos, 128, m0, kk, lane);
            v16h as = frag_a_load(tsin, 128, m0, kk, lane);
            v16h bx = frag_b_loadT(sxT, 136, kk, n0, lane);
            cr = wmma(ac, bx, cr);
            ci = wmma(as, bx, ci);
        }
        FragF fr, fi; fr.v = cr; fi.v = ci;
        size_t ob = ((size_t)bh * 65 + m0 + 8 * hi) * 768 + c0 + n0 + n;
        _Float16* yr = Yr + ob;
        _Float16* yi = Yi + ob;
        #pragma unroll
        for (int v = 0; v < 8; ++v) {
            nt_h(yr + (size_t)v * 768, (_Float16)fr.f[v]);
            nt_h(yi + (size_t)v * 768, (_Float16)fi.f[v]);
        }
    }
    // Nyquist tiles (m0 = 64): only row kw == 64 (v==0, hi==0) is stored
    for (int t = 64 + wave; t < 80; t += 8) {
        int n0 = (t - 64) * 16;
        v8f cr = {}; v8f ci = {};
        #pragma unroll
        for (int kk = 0; kk < 128; kk += 32) {
            v16h ac = frag_a_load(tcos, 128, 64, kk, lane);
            v16h as = frag_a_load(tsin, 128, 64, kk, lane);
            v16h bx = frag_b_loadT(sxT, 136, kk, n0, lane);
            cr = wmma(ac, bx, cr);
            ci = wmma(as, bx, ci);
        }
        FragF fr, fi; fr.v = cr; fi.v = ci;
        if (hi == 0) {
            size_t o = ((size_t)bh * 65 + 64) * 768 + c0 + n0 + n;
            nt_h(Yr + o, (_Float16)fr.f[0]);
            nt_h(Yi + o, (_Float16)fi.f[0]);
        }
    }
}

// =====================  K2/K4: column complex DFT (along H) =====================
__global__ __launch_bounds__(256) void k_col_fft(
    const _Float16* __restrict__ inr, const _Float16* __restrict__ ini,
    const _Float16* __restrict__ tcos, const _Float16* __restrict__ tsin,
    _Float16* __restrict__ outr, _Float16* __restrict__ outi) {
    __align__(16) __shared__ _Float16 syrT[16 * 136], syiT[16 * 136];   // 8.5 KB, [col][h]
    int wg = blockIdx.x;                                 // 12480 = 4 * 3120
    int qt = wg % 3120, b = wg / 3120;
    int q0 = qt * 16;
    int tid = threadIdx.x;
    const size_t QS = 49920;
    const _Float16* pr = inr + (size_t)b * 128 * QS + q0;
    const _Float16* pi = ini + (size_t)b * 128 * QS + q0;
    {   // 256 threads == 16 cols x 16 h-chunks of 8
        int col = tid & 15, h0 = (tid >> 4) * 8;
        V8HU a, c;
        #pragma unroll
        for (int e = 0; e < 8; ++e) {
            a.h[e] = pr[(size_t)(h0 + e) * QS + col];
            c.h[e] = pi[(size_t)(h0 + e) * QS + col];
        }
        *(v8h*)&syrT[col * 136 + h0] = a.v;
        *(v8h*)&syiT[col * 136 + h0] = c.v;
    }
    __syncthreads();
    int lane = tid & 31, wave = tid >> 5;
    int m0 = wave * 16;                                  // 8 waves cover kh 0..127
    v8f cr = {}; v8f ci = {};
    #pragma unroll
    for (int kk = 0; kk < 128; kk += 32) {
        v16h ac = frag_a_load(tcos, 128, m0, kk, lane);
        v16h as = frag_a_load(tsin, 128, m0, kk, lane);
        v16h br = frag_b_loadT(syrT, 136, kk, 0, lane);
        v16h bi = frag_b_loadT(syiT, 136, kk, 0, lane);
        v16h bin = frag_neg(bi);
        cr = wmma(ac, br, cr);
        cr = wmma(as, bin, cr);
        ci = wmma(ac, bi, ci);
        ci = wmma(as, br, ci);
    }
    FragF fr, fi; fr.v = cr; fi.v = ci;
    int n = lane & 15, hi = lane >> 4;
    size_t ob = (size_t)b * 128 * QS + (size_t)(m0 + 8 * hi) * QS + q0 + n;
    _Float16* qr = outr + ob;
    _Float16* qi = outi + ob;
    #pragma unroll
    for (int v = 0; v < 8; ++v) {
        nt_h(qr + (size_t)v * QS, (_Float16)fr.f[v]);
        nt_h(qi + (size_t)v * QS, (_Float16)fi.f[v]);
    }
}

// =====================  K3: fused complex block MLP =====================
__global__ __launch_bounds__(256) void k_mlp(
    const _Float16* __restrict__ Zr, const _Float16* __restrict__ Zi,
    const float* __restrict__ w1, const float* __restrict__ b1,
    const float* __restrict__ w2, const float* __restrict__ b2,
    _Float16* __restrict__ Or, _Float16* __restrict__ Oi) {
    __align__(16) __shared__ _Float16 sw1rT[96 * 104], sw1iT[96 * 104],
                                      sw2rT[96 * 104], sw2iT[96 * 104];   // 78 KB, [o][i]
    __align__(16) __shared__ _Float16 so1[8][2][16 * 96];                 // 48 KB
    int wg = blockIdx.x;                       // 2080 = 8 blocks * 260 pgroups
    int blk = wg % 8, pg = wg / 8;
    int tid = threadIdx.x;
    const float* g1r = w1 + (size_t)blk * 9216;
    const float* g1i = w1 + (size_t)(8 + blk) * 9216;
    const float* g2r = w2 + (size_t)blk * 9216;
    const float* g2i = w2 + (size_t)(8 + blk) * 9216;
    #pragma unroll
    for (int j = 0; j < 5; ++j) {              // 1152 b128 stores per matrix
        int idx = tid + j * 256;
        if (idx < 1152) {
            int o = idx / 12, i0 = (idx % 12) * 8;
            V8HU a1, c1, a2, c2;
            #pragma unroll
            for (int e = 0; e < 8; ++e) {
                int gi = (i0 + e) * 96 + o;
                a1.h[e] = (_Float16)g1r[gi];
                c1.h[e] = (_Float16)g1i[gi];
                a2.h[e] = (_Float16)g2r[gi];
                c2.h[e] = (_Float16)g2i[gi];
            }
            *(v8h*)&sw1rT[o * 104 + i0] = a1.v;
            *(v8h*)&sw1iT[o * 104 + i0] = c1.v;
            *(v8h*)&sw2rT[o * 104 + i0] = a2.v;
            *(v8h*)&sw2iT[o * 104 + i0] = c2.v;
        }
    }
    __syncthreads();
    int lane = tid & 31, wave = tid >> 5;
    int n = lane & 15, hi = lane >> 4;
    int p0 = pg * 128 + wave * 16;             // 16 points per wave, 33280 total
    const _Float16* ar_base = Zr + (size_t)p0 * 768 + blk * 96;
    const _Float16* ai_base = Zi + (size_t)p0 * 768 + blk * 96;
    // ---- layer 1: o1 = relu(Z*W1 + b1), complex ----
    #pragma unroll
    for (int ot = 0; ot < 6; ++ot) {
        int o0 = ot * 16;
        v8f c1r = {}; v8f c1i = {};
        #pragma unroll
        for (int kk = 0; kk < 96; kk += 32) {
            v16h axr = frag_a_load(ar_base, 768, 0, kk, lane);
            v16h axi = frag_a_load(ai_base, 768, 0, kk, lane);
            v16h axin = frag_neg(axi);
            v16h bw1r = frag_b_loadT(sw1rT, 104, kk, o0, lane);
            v16h bw1i = frag_b_loadT(sw1iT, 104, kk, o0, lane);
            c1r = wmma(axr, bw1r, c1r);
            c1r = wmma(axin, bw1i, c1r);
            c1i = wmma(axr, bw1i, c1i);
            c1i = wmma(axi, bw1r, c1i);
        }
        float bb1r = b1[blk * 96 + o0 + n];
        float bb1i = b1[768 + blk * 96 + o0 + n];
        FragF f1r, f1i; f1r.v = c1r; f1i.v = c1i;
        _Float16* s0 = &so1[wave][0][8 * hi * 96 + o0 + n];
        _Float16* s1 = &so1[wave][1][8 * hi * 96 + o0 + n];
        #pragma unroll
        for (int v = 0; v < 8; ++v) {
            float vr = f1r.f[v] + bb1r; if (vr < 0.f) vr = 0.f;
            float vi = f1i.f[v] + bb1i; if (vi < 0.f) vi = 0.f;
            s0[v * 96] = (_Float16)vr;
            s1[v * 96] = (_Float16)vi;
        }
    }
    __syncthreads();
    // ---- layer 2: o2 = o1*W2 + b2, complex; softshrink ----
    const _Float16* o1r = &so1[wave][0][0];
    const _Float16* o1i = &so1[wave][1][0];
    #pragma unroll
    for (int ot = 0; ot < 6; ++ot) {
        int o0 = ot * 16;
        v8f c2r = {}; v8f c2i = {};
        #pragma unroll
        for (int kk = 0; kk < 96; kk += 32) {
            v16h axr = frag_a_load(o1r, 96, 0, kk, lane);
            v16h axi = frag_a_load(o1i, 96, 0, kk, lane);
            v16h axin = frag_neg(axi);
            v16h bw2r = frag_b_loadT(sw2rT, 104, kk, o0, lane);
            v16h bw2i = frag_b_loadT(sw2iT, 104, kk, o0, lane);
            c2r = wmma(axr, bw2r, c2r);
            c2r = wmma(axin, bw2i, c2r);
            c2i = wmma(axr, bw2i, c2i);
            c2i = wmma(axi, bw2r, c2i);
        }
        float bb2r = b2[blk * 96 + o0 + n];
        float bb2i = b2[768 + blk * 96 + o0 + n];
        FragF f2r, f2i; f2r.v = c2r; f2i.v = c2i;
        size_t ob = (size_t)(p0 + 8 * hi) * 768 + blk * 96 + o0 + n;
        _Float16* por = Or + ob;
        _Float16* poi = Oi + ob;
        #pragma unroll
        for (int v = 0; v < 8; ++v) {
            float vr = f2r.f[v] + bb2r;
            float vi = f2i.f[v] + bb2i;
            vr = (vr > LAM) ? vr - LAM : ((vr < -LAM) ? vr + LAM : 0.f);
            vi = (vi > LAM) ? vi - LAM : ((vi < -LAM) ? vi + LAM : 0.f);
            nt_h(por + (size_t)v * 768, (_Float16)vr);
            nt_h(poi + (size_t)v * 768, (_Float16)vi);
        }
    }
}

// =====================  K5: inverse row rFFT + residual =====================
__global__ __launch_bounds__(256) void k_row_ifft(
    const _Float16* __restrict__ Zr, const _Float16* __restrict__ Zi,
    const _Float16* __restrict__ icos, const _Float16* __restrict__ isin,
    const float* __restrict__ x, float* __restrict__ out) {
    __align__(16) __shared__ _Float16 szrT[256 * 104], sziT[256 * 104];   // 104 KB, [c][k]
    int wg = blockIdx.x;                                 // 1536
    int cg = wg % 3, bh = wg / 3;
    int c0 = cg * 256;
    int tid = threadIdx.x;
    const _Float16* pr = Zr + (size_t)bh * 49920 + c0;
    const _Float16* pi = Zi + (size_t)bh * 49920 + c0;
    for (int j = 0; j < 12; ++j) {            // 256 cc x 12 k-chunks of 8
        int idx = tid + j * 256;
        int cc = idx & 255, k0c = (idx >> 8) * 8;
        V8HU a, c;
        #pragma unroll
        for (int e = 0; e < 8; ++e) {
            int k = k0c + e;
            a.h[e] = (k < 65) ? pr[(size_t)k * 768 + cc] : (_Float16)0.f;
            c.h[e] = (k < 65) ? pi[(size_t)k * 768 + cc] : (_Float16)0.f;
        }
        *(v8h*)&szrT[cc * 104 + k0c] = a.v;
        *(v8h*)&sziT[cc * 104 + k0c] = c.v;
    }
    __syncthreads();
    int lane = tid & 31, wave = tid >> 5;
    int n = lane & 15, hi = lane >> 4;
    for (int t = wave; t < 128; t += 8) {     // 8 w-tiles x 16 c-tiles
        int m0 = (t >> 4) * 16, n0 = (t & 15) * 16;
        v8f acc = {};
        #pragma unroll
        for (int kk = 0; kk < 96; kk += 32) {
            v16h ac = frag_a_load(icos, 96, m0, kk, lane);
            v16h as = frag_a_load(isin, 96, m0, kk, lane);
            v16h br = frag_b_loadT(szrT, 104, kk, n0, lane);
            v16h bi = frag_b_loadT(sziT, 104, kk, n0, lane);
            acc = wmma(ac, br, acc);
            acc = wmma(as, bi, acc);
        }
        FragF f; f.v = acc;
        size_t ob = ((size_t)bh * 128 + m0 + 8 * hi) * 768 + c0 + n0 + n;
        const float* px = x + ob;
        float* po = out + ob;
        #pragma unroll
        for (int v = 0; v < 8; ++v)
            nt_f(po + (size_t)v * 768, f.f[v] + px[(size_t)v * 768]);
    }
}

extern "C" void kernel_launch(void* const* d_in, const int* in_sizes, int n_in,
                              void* d_out, int out_size, void* d_ws, size_t ws_size,
                              hipStream_t stream) {
    (void)in_sizes; (void)n_in; (void)out_size; (void)ws_size;
    const float* x  = (const float*)d_in[0];
    const float* w1 = (const float*)d_in[1];
    const float* b1 = (const float*)d_in[2];
    const float* w2 = (const float*)d_in[3];
    const float* b2 = (const float*)d_in[4];
    float* out = (float*)d_out;

    char* ws = (char*)d_ws;
    _Float16* tcos  = (_Float16*)(ws);                 // 128x128
    _Float16* tsinf = (_Float16*)(ws + 32768);         // -sin (forward)
    _Float16* tsini = (_Float16*)(ws + 65536);         // +sin (inverse)
    _Float16* icos  = (_Float16*)(ws + 98304);         // 128x96
    _Float16* isin  = (_Float16*)(ws + 122880);
    size_t off = 147456;
    const size_t NBUF = (size_t)4 * 128 * 65 * 768;    // freq-domain elements
    _Float16* Yr = (_Float16*)(ws + off); off += NBUF * 2;
    _Float16* Yi = (_Float16*)(ws + off); off += NBUF * 2;
    _Float16* Zr = (_Float16*)(ws + off); off += NBUF * 2;
    _Float16* Zi = (_Float16*)(ws + off); off += NBUF * 2;

    k_init_tables<<<112, 256, 0, stream>>>(tcos, tsinf, tsini, icos, isin);
    k_row_fft<<<1536, 256, 0, stream>>>(x, tcos, tsinf, Yr, Yi);
    k_col_fft<<<12480, 256, 0, stream>>>(Yr, Yi, tcos, tsinf, Zr, Zi);
    k_mlp<<<2080, 256, 0, stream>>>(Zr, Zi, w1, b1, w2, b2, Yr, Yi);
    k_col_fft<<<12480, 256, 0, stream>>>(Yr, Yi, tcos, tsini, Zr, Zi);
    k_row_ifft<<<1536, 256, 0, stream>>>(Zr, Zi, icos, isin, x, out);
}
